// BERT_10153302687798
// MI455X (gfx1250) — matrix operand order repaired and checked by reading, not compiled, not run on hardware
//
#include <hip/hip_runtime.h>
#include <hip/hip_bf16.h>

typedef __attribute__((ext_vector_type(16))) _Float16 v16h;
typedef __attribute__((ext_vector_type(8)))  _Float16 v8h;
typedef __attribute__((ext_vector_type(8)))  float    v8f;

#define D_   768
#define DF_  3072
#define H_   12
#define HD_  64
#define L_   12
#define B_   8
#define T_   512
#define V_   30522
#define M_   4096          // B*T rows
#define VPAD 30592         // ceil(V/128)*128

// ---------------- CDNA5 TDM availability (device pass only) ----------------
#if defined(__gfx1250__) && __has_builtin(__builtin_amdgcn_tensor_load_to_lds) && \
    __has_builtin(__builtin_amdgcn_s_wait_tensorcnt)
#define USE_TDM 1
#else
#define USE_TDM 0
#endif

#if USE_TDM
typedef __attribute__((ext_vector_type(4))) unsigned int u32x4;
typedef __attribute__((ext_vector_type(8))) int          i32x8;
typedef __attribute__((ext_vector_type(4))) int          i32x4;

// Issue one TDM descriptor moving a [128 x 32] f16 tile (row stride = lda
// elements) from global to LDS, inserting 4 dwords of pad every 16 dwords so
// the LDS row stride becomes LDT=40 halves. See CDNA5 ISA.
__device__ __forceinline__ void tdm_load_a(const _Float16* gsrc, unsigned lds_off,
                                           int lda, int rem_k, int rem_m) {
  unsigned long long ga = (unsigned long long)(size_t)gsrc;
  u32x4 g0;
  g0[0] = 1u;                                   // count=1, user mode, no gather
  g0[1] = lds_off;                              // D#.lds_addr (bytes)
  g0[2] = (unsigned)(ga & 0xffffffffu);         // global_addr[31:0]  (bits 95:64)
  g0[3] = (unsigned)((ga >> 32) & 0x01ffffffu)  // global_addr[56:32] (bits 120:96)
        | (2u << 30);                           // type=2 ("image")
  i32x8 g1;
  unsigned td0 = (unsigned)rem_k;               // tensor_dim0 (elements)
  unsigned td1 = (unsigned)rem_m;               // tensor_dim1 (rows)
  unsigned long long st0 = (unsigned long long)lda;  // tensor_dim0_stride
  g1[0] = (int)((1u << 16)                      // data_size = 2 bytes
              | (1u << 20)                      // pad_enable
              | (3u << 22)                      // pad_interval: 16 dwords
              | (3u << 25));                    // pad_amount:   4 dwords
  g1[1] = (int)((td0 & 0xffffu) << 16);                         // tensor_dim0[15:0]
  g1[2] = (int)((td0 >> 16) | ((td1 & 0xffffu) << 16));         // dim0 hi | dim1 lo
  g1[3] = (int)((td1 >> 16) | (32u << 16));                     // dim1 hi | tile_dim0=32
  g1[4] = (int)(128u);                                          // tile_dim1=128, tile_dim2=0
  g1[5] = (int)(st0 & 0xffffffffu);                             // dim0_stride[31:0]
  g1[6] = (int)((st0 >> 32) & 0xffffu);                         // dim0_stride[47:32]
  g1[7] = 0;
  i32x4 z4 = {0, 0, 0, 0};                                      // 2-D: groups 2/3 unused
  i32x8 z8 = {0, 0, 0, 0, 0, 0, 0, 0};
  __builtin_amdgcn_tensor_load_to_lds(g0, g1, z4, z4, z8, 0);   // 6-arg toolchain form
}
#endif

__device__ __forceinline__ void tdm_wait() {
#if USE_TDM
  __builtin_amdgcn_s_wait_tensorcnt(0);
#endif
}

// ------------------------------------------------------------------
// Embedding sum: x = word_emb[ids] + pos_emb[t] + type_emb[tty]
// ------------------------------------------------------------------
__global__ void embed_kernel(const int* __restrict__ ids, const int* __restrict__ tty,
                             const float* __restrict__ we, const float* __restrict__ pe,
                             const float* __restrict__ te, float* __restrict__ x) {
  int idx = blockIdx.x * blockDim.x + threadIdx.x;
  if (idx >= M_ * D_) return;
  int row = idx / D_, d = idx - row * D_;
  int t = row & (T_ - 1);
  x[idx] = we[(size_t)ids[row] * D_ + d] + pe[t * D_ + d] + te[(size_t)tty[row] * D_ + d];
}

// ------------------------------------------------------------------
// LayerNorm over D=768, one block per row; writes f32 and/or f16
// ------------------------------------------------------------------
__global__ __launch_bounds__(256) void ln_kernel(const float* __restrict__ in,
                                                 float* __restrict__ outF,
                                                 _Float16* __restrict__ outH,
                                                 const float* __restrict__ g,
                                                 const float* __restrict__ bt) {
  __shared__ float s1[256], s2[256];
  int row = blockIdx.x;
  const float* r = in + (size_t)row * D_;
  float v[3]; float s = 0.f, sq = 0.f;
#pragma unroll
  for (int i = 0; i < 3; ++i) {
    v[i] = r[threadIdx.x + i * 256];
    s += v[i]; sq += v[i] * v[i];
  }
  s1[threadIdx.x] = s; s2[threadIdx.x] = sq;
  __syncthreads();
  for (int st = 128; st > 0; st >>= 1) {
    if (threadIdx.x < st) {
      s1[threadIdx.x] += s1[threadIdx.x + st];
      s2[threadIdx.x] += s2[threadIdx.x + st];
    }
    __syncthreads();
  }
  float mean = s1[0] * (1.f / D_);
  float var  = s2[0] * (1.f / D_) - mean * mean;
  float rstd = rsqrtf(var + 1e-5f);
#pragma unroll
  for (int i = 0; i < 3; ++i) {
    int c = threadIdx.x + i * 256;
    float o = (v[i] - mean) * rstd * g[c] + bt[c];
    if (outF) outF[(size_t)row * D_ + c] = o;
    if (outH) outH[(size_t)row * D_ + c] = (_Float16)o;
  }
}

// ------------------------------------------------------------------
// Weight conversion fp32 -> f16 with optional column padding to ldb
// ------------------------------------------------------------------
__global__ void cvt_w_kernel(const float* __restrict__ src, _Float16* __restrict__ dst,
                             int K, int N, int ldb) {
  int idx = blockIdx.x * blockDim.x + threadIdx.x;
  if (idx >= K * ldb) return;
  int k = idx / ldb, n = idx - k * ldb;
  dst[idx] = (n < N) ? (_Float16)src[(size_t)k * N + n] : (_Float16)0.f;
}

// Head weight: word_emb [V,D] transposed to [D, VPAD] f16 with zero pad
__global__ void cvt_wT_kernel(const float* __restrict__ src, _Float16* __restrict__ dst) {
  int idx = blockIdx.x * blockDim.x + threadIdx.x;
  if (idx >= D_ * VPAD) return;
  int d = idx / VPAD, n = idx - d * VPAD;
  dst[idx] = (n < V_) ? (_Float16)src[(size_t)n * D_ + d] : (_Float16)0.f;
}

__global__ void cvt_f32_f16(const float* __restrict__ s, _Float16* __restrict__ d, int n) {
  int idx = blockIdx.x * blockDim.x + threadIdx.x;
  if (idx < n) d[idx] = (_Float16)s[idx];
}

// ------------------------------------------------------------------
// WMMA GEMM: C[M,N] = A[M,K](f16) @ B[K,N](f16, ldb-padded) + bias
//            (+ residual) (+ GELU), out f32 or f16
// 128x128x32 block tiles, 8 waves, 32x64 per wave (2x4 WMMA tiles).
// Double-buffered LDS; A tile staged by the Tensor Data Mover (wave 0),
// B tile staged transposed by all waves.
// ------------------------------------------------------------------
#define BM  128
#define BN  128
#define BK  32
#define LDT 40   // LDS row stride in halves (80B, multiple of 16B)

__device__ __forceinline__ v16h load_frag_a(const _Float16* p) {
  // A 16-bit 16x32 layout: per lane, K = kb..kb+7 then kb+16..kb+23
  v8h lo = *(const v8h*)p;
  v8h hi = *(const v8h*)(p + 16);
  v16h f;
#pragma unroll
  for (int i = 0; i < 8; ++i) { f[i] = lo[i]; f[i + 8] = hi[i]; }
  return f;
}

__device__ __forceinline__ v16h load_frag_b(const _Float16* p) {
  // B 16-bit 32x16 layout: per lane, K = kb..kb+7 then kb+16..kb+23
  v8h lo = *(const v8h*)p;
  v8h hi = *(const v8h*)(p + 16);
  v16h f;
#pragma unroll
  for (int i = 0; i < 8; ++i) { f[i] = lo[i]; f[i + 8] = hi[i]; }
  return f;
}

template<bool OUTH, bool GELU, bool RES>
__global__ __launch_bounds__(256) void gemm_wmma(
    const _Float16* __restrict__ A, int lda,
    const _Float16* __restrict__ Bw, int ldb,
    const float* __restrict__ bias,
    const float* __restrict__ res,
    float* __restrict__ outF, _Float16* __restrict__ outH,
    int N, int K) {
  __shared__ _Float16 As[2][BM * LDT];     // [row][k], row-major
  __shared__ _Float16 Bt[2][BN * LDT];     // [col][k], transposed tile
  const int bm   = blockIdx.y * BM;
  const int bn   = blockIdx.x * BN;
  const int tid  = threadIdx.x;
  const int lane = tid & 31;
  const int wave = tid >> 5;
  const int wm   = (wave & 3) * 32;        // wave M offset in block tile
  const int wn   = (wave >> 2) * 64;       // wave N offset in block tile
  const int lr   = lane & 15;
  const int kbA  = (lane < 16) ? 0 : 8;
  const int kbB  = (lane < 16) ? 0 : 8;

  v8f acc[2][4];
#pragma unroll
  for (int mi = 0; mi < 2; ++mi)
#pragma unroll
    for (int ni = 0; ni < 4; ++ni)
#pragma unroll
      for (int e = 0; e < 8; ++e) acc[mi][ni][e] = 0.f;

#if USE_TDM
  const unsigned as_base = (unsigned)(size_t)(&As[0][0]);
#endif

  // Stage one K-tile (A via TDM or fallback, B transposed manually)
  auto stage = [&](int buf, int kt) {
    // B tile transposed: 32 k-rows x 128 cols -> Bt[buf][col][k]
#pragma unroll
    for (int i = 0; i < 2; ++i) {
      int idx = tid + i * 256;
      int r = idx >> 4;               // k within tile
      int c = (idx & 15) << 3;        // n within tile
      v8h gb = *(const v8h*)(Bw + (size_t)(kt + r) * ldb + bn + c);
#pragma unroll
      for (int j = 0; j < 8; ++j) Bt[buf][(c + j) * LDT + r] = gb[j];
    }
#if USE_TDM
    if (wave == 0)
      tdm_load_a(A + (size_t)bm * lda + kt,
                 as_base + (unsigned)(buf * (BM * LDT * 2)),
                 lda, K - kt, M_ - bm);
#else
    // A tile: 128 rows x 32 halves, vectorized 16B loads/stores
#pragma unroll
    for (int i = 0; i < 2; ++i) {
      int idx = tid + i * 256;
      int r = idx >> 2;
      int c = (idx & 3) << 3;
      *(uint4*)&As[buf][r * LDT + c] =
          *(const uint4*)(A + (size_t)(bm + r) * lda + kt + c);
    }
#endif
  };

  stage(0, 0);
  tdm_wait();
  __syncthreads();

  const int nk = K / BK;
  for (int t = 0; t < nk; ++t) {
    const int cur = t & 1;
    if (t + 1 < nk) stage(cur ^ 1, (t + 1) * BK);

    v16h af[2], bf[4];
#pragma unroll
    for (int mi = 0; mi < 2; ++mi)
      af[mi] = load_frag_a(&As[cur][(wm + mi * 16 + lr) * LDT + kbA]);
#pragma unroll
    for (int ni = 0; ni < 4; ++ni)
      bf[ni] = load_frag_b(&Bt[cur][(wn + ni * 16 + lr) * LDT + kbB]);
#pragma unroll
    for (int mi = 0; mi < 2; ++mi)
#pragma unroll
      for (int ni = 0; ni < 4; ++ni)
        acc[mi][ni] = __builtin_amdgcn_wmma_f32_16x16x32_f16(
            false, af[mi], false, bf[ni], (short)0, acc[mi][ni], false, false);

    tdm_wait();
    __syncthreads();
  }

  // Epilogue. C/D layout: lanes 0-15 -> M=r, lanes 16-31 -> M=r+8, N=lane&15
  const int rbase = bm + wm + ((lane < 16) ? 0 : 8);
#pragma unroll
  for (int mi = 0; mi < 2; ++mi) {
#pragma unroll
    for (int ni = 0; ni < 4; ++ni) {
      int col = bn + wn + ni * 16 + lr;
      if (col < N) {
        float bv = bias ? bias[col] : 0.f;
#pragma unroll
        for (int r2 = 0; r2 < 8; ++r2) {
          int grow = rbase + mi * 16 + r2;
          float v = acc[mi][ni][r2] + bv;
          if (RES) v += res[(size_t)grow * N + col];
          if (GELU) {
            float u = 0.7978845608f * (v + 0.044715f * v * v * v);
            v = 0.5f * v * (1.f + tanhf(u));
          }
          if (OUTH) outH[(size_t)grow * N + col] = (_Float16)v;
          else      outF[(size_t)grow * N + col] = v;
        }
      }
    }
  }
}

// ------------------------------------------------------------------
// Fused attention: online-softmax, one thread per query row,
// K/V streamed through LDS in 64-key chunks. qkv16 is [M, 3D] f16.
// ------------------------------------------------------------------
#define CHNK 64
__global__ __launch_bounds__(128) void attn_kernel(const _Float16* __restrict__ qkv,
                                                   const float* __restrict__ mask,
                                                   _Float16* __restrict__ y) {
  __shared__ float Ks[CHNK][HD_ + 1];
  __shared__ float Vs[CHNK][HD_ + 1];
  const int bh = blockIdx.y;
  const int b = bh / H_, h = bh - b * H_;
  const int q = blockIdx.x * 128 + threadIdx.x;
  const int row = b * T_ + q;

  const _Float16* qp = qkv + (size_t)row * (3 * D_) + h * HD_;
  float qr[HD_];
#pragma unroll
  for (int d = 0; d < HD_; ++d) qr[d] = (float)qp[d] * 0.125f;  // 1/sqrt(64)

  float m = -3.0e38f, l = 0.f, acc[HD_];
#pragma unroll
  for (int d = 0; d < HD_; ++d) acc[d] = 0.f;

  for (int kc = 0; kc < T_; kc += CHNK) {
    __syncthreads();
    {
      int kr   = threadIdx.x >> 1;                 // 0..63
      int half = (threadIdx.x & 1) * 32;
      const _Float16* kp = qkv + (size_t)(b * T_ + kc + kr) * (3 * D_) + D_     + h * HD_ + half;
      const _Float16* vp = qkv + (size_t)(b * T_ + kc + kr) * (3 * D_) + 2 * D_ + h * HD_ + half;
#pragma unroll
      for (int d = 0; d < 32; ++d) {
        Ks[kr][half + d] = (float)kp[d];
        Vs[kr][half + d] = (float)vp[d];
      }
    }
    __syncthreads();
    for (int j = 0; j < CHNK; ++j) {
      float s = 0.f;
#pragma unroll
      for (int d = 0; d < HD_; ++d) s += qr[d] * Ks[j][d];
      s += (1.f - mask[b * T_ + kc + j]) * (-10000.f);
      float mn   = fmaxf(m, s);
      float corr = __expf(m - mn);
      float p    = __expf(s - mn);
      l = l * corr + p;
#pragma unroll
      for (int d = 0; d < HD_; ++d) acc[d] = acc[d] * corr + p * Vs[j][d];
      m = mn;
    }
  }
  float inv = 1.f / l;
  _Float16* yo = y + (size_t)row * D_ + h * HD_;
#pragma unroll
  for (int d = 0; d < HD_; ++d) yo[d] = (_Float16)(acc[d] * inv);
}

// ------------------------------------------------------------------
extern "C" void kernel_launch(void* const* d_in, const int* in_sizes, int n_in,
                              void* d_out, int out_size, void* d_ws, size_t ws_size,
                              hipStream_t stream) {
  (void)in_sizes; (void)n_in; (void)out_size; (void)ws_size;
  const int*   ids   = (const int*)d_in[0];
  const int*   tty   = (const int*)d_in[1];
  const float* mask  = (const float*)d_in[2];
  const float* we    = (const float*)d_in[3];
  const float* pe    = (const float*)d_in[4];
  const float* te    = (const float*)d_in[5];
  const float* elng  = (const float*)d_in[6];
  const float* elnb  = (const float*)d_in[7];
  const float* ln1g  = (const float*)d_in[8];
  const float* ln1b  = (const float*)d_in[9];
  const float* wqkv  = (const float*)d_in[10];
  const float* bqkv  = (const float*)d_in[11];
  const float* wproj = (const float*)d_in[12];
  const float* bproj = (const float*)d_in[13];
  const float* ln2g  = (const float*)d_in[14];
  const float* ln2b  = (const float*)d_in[15];
  const float* wfc   = (const float*)d_in[16];
  const float* bfc   = (const float*)d_in[17];
  const float* wfc2  = (const float*)d_in[18];
  const float* bfc2  = (const float*)d_in[19];

  // Workspace carve-out (~117 MB total), 256B aligned slabs
  char* ws = (char*)d_ws;
  size_t off = 0;
  auto alloc = [&](size_t bytes) -> char* {
    char* p = ws + off;
    off += (bytes + 255) & ~(size_t)255;
    return p;
  };
  float*    x     = (float*)   alloc((size_t)M_ * D_ * 4);       // residual stream f32
  _Float16* h16   = (_Float16*)alloc((size_t)M_ * D_ * 2);       // LN output / head input
  _Float16* qkv16 = (_Float16*)alloc((size_t)M_ * 3 * D_ * 2);   // QKV GEMM out
  _Float16* y16   = (_Float16*)alloc((size_t)M_ * D_ * 2);       // attention out
  _Float16* ff16  = (_Float16*)alloc((size_t)M_ * DF_ * 2);      // GELU(FC1) out
  _Float16* wbuf  = (_Float16*)alloc((size_t)D_ * VPAD * 2);     // per-GEMM f16 weight slab

  // Embedding + LN (in-place on x)
  embed_kernel<<<(M_ * D_ + 255) / 256, 256, 0, stream>>>(ids, tty, we, pe, te, x);
  ln_kernel<<<M_, 256, 0, stream>>>(x, x, nullptr, elng, elnb);

  for (int l = 0; l < L_; ++l) {
    // LN1 -> h16
    ln_kernel<<<M_, 256, 0, stream>>>(x, nullptr, h16, ln1g + l * D_, ln1b + l * D_);
    // qkv16 = h16 @ wqkv_l + bqkv_l
    cvt_w_kernel<<<(D_ * 3 * D_ + 255) / 256, 256, 0, stream>>>(
        wqkv + (size_t)l * D_ * 3 * D_, wbuf, D_, 3 * D_, 3 * D_);
    gemm_wmma<true, false, false><<<dim3(3 * D_ / BN, M_ / BM), 256, 0, stream>>>(
        h16, D_, wbuf, 3 * D_, bqkv + (size_t)l * 3 * D_, nullptr,
        nullptr, qkv16, 3 * D_, D_);
    // y16 = softmax(q kᵀ/√d + mask) v
    attn_kernel<<<dim3(T_ / 128, B_ * H_), 128, 0, stream>>>(qkv16, mask, y16);
    // x += y16 @ wproj_l + bproj_l
    cvt_w_kernel<<<(D_ * D_ + 255) / 256, 256, 0, stream>>>(
        wproj + (size_t)l * D_ * D_, wbuf, D_, D_, D_);
    gemm_wmma<false, false, true><<<dim3(D_ / BN, M_ / BM), 256, 0, stream>>>(
        y16, D_, wbuf, D_, bproj + (size_t)l * D_, x, x, nullptr, D_, D_);
    // LN2 -> h16
    ln_kernel<<<M_, 256, 0, stream>>>(x, nullptr, h16, ln2g + l * D_, ln2b + l * D_);
    // ff16 = gelu(h16 @ wfc_l + bfc_l)
    cvt_w_kernel<<<(D_ * DF_ + 255) / 256, 256, 0, stream>>>(
        wfc + (size_t)l * D_ * DF_, wbuf, D_, DF_, DF_);
    gemm_wmma<true, true, false><<<dim3(DF_ / BN, M_ / BM), 256, 0, stream>>>(
        h16, D_, wbuf, DF_, bfc + (size_t)l * DF_, nullptr, nullptr, ff16, DF_, D_);
    // x += ff16 @ wfc2_l + bfc2_l
    cvt_w_kernel<<<(DF_ * D_ + 255) / 256, 256, 0, stream>>>(
        wfc2 + (size_t)l * DF_ * D_, wbuf, DF_, D_, D_);
    gemm_wmma<false, false, true><<<dim3(D_ / BN, M_ / BM), 256, 0, stream>>>(
        ff16, DF_, wbuf, D_, bfc2 + (size_t)l * D_, x, x, nullptr, D_, DF_);
  }

  // MLM head: logits = x @ word_embᵀ  (f16 inputs, f32 out to d_out)
  cvt_f32_f16<<<(M_ * D_ + 255) / 256, 256, 0, stream>>>(x, h16, M_ * D_);
  cvt_wT_kernel<<<(D_ * VPAD + 255) / 256, 256, 0, stream>>>(we, wbuf);
  gemm_wmma<false, false, false><<<dim3((V_ + BN - 1) / BN, M_ / BM), 256, 0, stream>>>(
      h16, D_, wbuf, VPAD, nullptr, nullptr, (float*)d_out, nullptr, V_, D_);
}
